// PrimeQwen2MLP_50663434223830
// MI455X (gfx1250) — matrix operand order repaired and checked by reading, not compiled, not run on hardware
//
#include <hip/hip_runtime.h>
#include <hip/hip_bf16.h>

#define HIDDEN 2048
#define INTER  5504
#define MTOK   8192   // 2 * 4096 tokens

typedef __attribute__((ext_vector_type(16))) _Float16 v16h;
typedef __attribute__((ext_vector_type(8)))  _Float16 v8h;
typedef __attribute__((ext_vector_type(4)))  _Float16 v4h;
typedef __attribute__((ext_vector_type(8)))  float    v8f;
typedef __attribute__((ext_vector_type(4)))  float    v4f;
typedef __attribute__((ext_vector_type(4)))  int      v4i;

// ---------------------------------------------------------------------------
// Dequant: 4-bit codes (stored as int32) -> f16 weights via 16-entry LUT.
// ---------------------------------------------------------------------------
__global__ __launch_bounds__(256)
void lut_dequant_f16(const int* __restrict__ idx, const float* __restrict__ lut,
                     _Float16* __restrict__ out, int n4) {
    __shared__ _Float16 slut[16];
    if (threadIdx.x < 16) slut[threadIdx.x] = (_Float16)lut[threadIdx.x];
    __syncthreads();
    int i = blockIdx.x * blockDim.x + threadIdx.x;
    if (i < n4) {
        v4i c = ((const v4i*)idx)[i];
        v4h o;
        o.x = slut[c.x & 15];
        o.y = slut[c.y & 15];
        o.z = slut[c.z & 15];
        o.w = slut[c.w & 15];
        ((v4h*)out)[i] = o;
    }
}

// ---------------------------------------------------------------------------
// x: f32 -> f16
// ---------------------------------------------------------------------------
__global__ __launch_bounds__(256)
void f32_to_f16(const float* __restrict__ in, _Float16* __restrict__ out, int n4) {
    int i = blockIdx.x * blockDim.x + threadIdx.x;
    if (i < n4) {
        v4f v = ((const v4f*)in)[i];
        v4h o;
        o.x = (_Float16)v.x; o.y = (_Float16)v.y;
        o.z = (_Float16)v.z; o.w = (_Float16)v.w;
        ((v4h*)out)[i] = o;
    }
}

// ---------------------------------------------------------------------------
// WMMA GEMM: C[M,N] = A[M,K] * B[N,K]^T, double-buffered LDS pipeline.
// Block tile 128x64, BK=32, 8 wave32 as 4(M) x 2(N), wave tile 32x32.
// FUSED: two weight matrices (gate/up) + SiLU epilogue -> f16 h.
// ---------------------------------------------------------------------------
template <bool FUSED>
__global__ __launch_bounds__(256)
void ghost_gemm(const _Float16* __restrict__ A,
                const _Float16* __restrict__ B0,
                const _Float16* __restrict__ B1,
                void* __restrict__ Cout,
                int M, int N, int K) {
    constexpr int BM = 128, BN = 64, BK = 32;
    constexpr int LDS_K = BK + 8;   // 80B row stride, bank-friendly

    __shared__ _Float16 sA [2][BM * LDS_K];
    __shared__ _Float16 sB0[2][BN * LDS_K];
    __shared__ _Float16 sB1[FUSED ? 2 : 1][BN * LDS_K];

    const int tid  = threadIdx.x;
    const int lane = tid & 31;
    const int wave = tid >> 5;       // 0..7
    const int wm   = wave >> 1;      // 0..3  -> 32-row slab
    const int wn   = wave & 1;       // 0..1  -> 32-col slab
    const int lrow = lane & 15;
    const int lhi  = lane >> 4;

    const int mBase = blockIdx.y * BM;
    const int nBase = blockIdx.x * BN;

    v8f accG[2][2] = {};
    v8f accU[2][2] = {};

    // Per-thread staging coordinates (tiles fully in-bounds by construction).
    const int ar = tid >> 1, ac = (tid & 1) * 16;   // A: 128 x 32 halves
    const int br = tid >> 2, bc = (tid & 3) * 8;    // B: 64 x 32 halves

    const _Float16* aPtr  = A  + (size_t)(mBase + ar) * K + ac;
    const _Float16* b0Ptr = B0 + (size_t)(nBase + br) * K + bc;
    const _Float16* b1Ptr = FUSED ? (B1 + (size_t)(nBase + br) * K + bc) : B0;

    // LDS write/read offsets
    _Float16* const swA0 = &sA[0][ar * LDS_K + ac];
    _Float16* const swB0 = &sB0[0][br * LDS_K + bc];
    _Float16* const swB1 = FUSED ? &sB1[0][br * LDS_K + bc] : nullptr;
    constexpr int BUFA  = BM * LDS_K;   // half-elements per A buffer
    constexpr int BUFB  = BN * LDS_K;

    // ---- prologue: stage tile 0 ----
    {
        v8h a0 = *(const v8h*)(aPtr);
        v8h a1 = *(const v8h*)(aPtr + 8);
        v8h b0 = *(const v8h*)(b0Ptr);
        *(v8h*)(swA0)     = a0;
        *(v8h*)(swA0 + 8) = a1;
        *(v8h*)(swB0)     = b0;
        if (FUSED) {
            v8h b1 = *(const v8h*)(b1Ptr);
            *(v8h*)(swB1) = b1;
        }
    }
    __syncthreads();

    int cur = 0;
    for (int k0 = 0; k0 < K; k0 += BK) {
        const bool hasNext = (k0 + BK) < K;

        // ---- prefetch next tile into registers ----
        v8h pa0, pa1, pb0, pb1;
        if (hasNext) {
            const int kn = k0 + BK;
            pa0 = *(const v8h*)(aPtr + kn);
            pa1 = *(const v8h*)(aPtr + kn + 8);
            pb0 = *(const v8h*)(b0Ptr + kn);
            if (FUSED) pb1 = *(const v8h*)(b1Ptr + kn);
        }

        // ---- fragments from LDS buffer `cur` ----
        const _Float16* bufA  = sA[cur];
        const _Float16* bufB0 = sB0[cur];
        const _Float16* bufB1 = FUSED ? sB1[cur] : nullptr;

        v16h aF[2], b0F[2], b1F[2];
#pragma unroll
        for (int i = 0; i < 2; ++i) {
            // A 16x32 f16: lane-group g holds contiguous K chunk [16g, 16g+16)
            const _Float16* ap = bufA + (wm * 32 + i * 16 + lrow) * LDS_K + lhi * 16;
            v8h lo = *(const v8h*)ap;
            v8h hi = *(const v8h*)(ap + 8);
            aF[i] = __builtin_shufflevector(lo, hi, 0, 1, 2, 3, 4, 5, 6, 7,
                                                    8, 9, 10, 11, 12, 13, 14, 15);
        }
#pragma unroll
        for (int j = 0; j < 2; ++j) {
            // B 32x16 f16: lane-group g holds contiguous K chunk [16g, 16g+16)
            const _Float16* bp = bufB0 + (wn * 32 + j * 16 + lrow) * LDS_K + lhi * 16;
            v8h lo = *(const v8h*)bp;
            v8h hi = *(const v8h*)(bp + 8);
            b0F[j] = __builtin_shufflevector(lo, hi, 0, 1, 2, 3, 4, 5, 6, 7,
                                                     8, 9, 10, 11, 12, 13, 14, 15);
            if (FUSED) {
                const _Float16* cp = bufB1 + (wn * 32 + j * 16 + lrow) * LDS_K + lhi * 16;
                v8h lo1 = *(const v8h*)cp;
                v8h hi1 = *(const v8h*)(cp + 8);
                b1F[j] = __builtin_shufflevector(lo1, hi1, 0, 1, 2, 3, 4, 5, 6, 7,
                                                           8, 9, 10, 11, 12, 13, 14, 15);
            }
        }

        // ---- matrix cores ----
#pragma unroll
        for (int i = 0; i < 2; ++i)
#pragma unroll
            for (int j = 0; j < 2; ++j) {
                accG[i][j] = __builtin_amdgcn_wmma_f32_16x16x32_f16(
                    false, aF[i], false, b0F[j], (short)0, accG[i][j], false, false);
                if (FUSED)
                    accU[i][j] = __builtin_amdgcn_wmma_f32_16x16x32_f16(
                        false, aF[i], false, b1F[j], (short)0, accU[i][j], false, false);
            }

        // ---- commit prefetch into the other buffer; single barrier ----
        if (hasNext) {
            const int nxt = (cur ^ 1);
            *(v8h*)(swA0 + nxt * BUFA)     = pa0;
            *(v8h*)(swA0 + nxt * BUFA + 8) = pa1;
            *(v8h*)(swB0 + nxt * BUFB)     = pb0;
            if (FUSED) *(v8h*)(swB1 + nxt * BUFB) = pb1;
            __syncthreads();
            cur ^= 1;
        }
    }

    // ---- epilogue: C layout VGPR r, lane l -> M = r + 8*(l>>4), N = l&15 ----
#pragma unroll
    for (int i = 0; i < 2; ++i)
#pragma unroll
        for (int j = 0; j < 2; ++j) {
            const int rowB = mBase + wm * 32 + i * 16 + lhi * 8;
            const int col  = nBase + wn * 32 + j * 16 + lrow;
            int off = rowB * N + col;               // < 2^31, int math
            if (FUSED) {
                _Float16* O = (_Float16*)Cout;
#pragma unroll
                for (int r = 0; r < 8; ++r) {
                    float g = accG[i][j][r];
                    float u = accU[i][j][r];
                    // silu(g)*u with fast hardware reciprocal (v_rcp_f32)
                    float sig = __builtin_amdgcn_rcpf(1.0f + __expf(-g));
                    O[off + r * N] = (_Float16)(g * sig * u);
                }
            } else {
                float* O = (float*)Cout;
#pragma unroll
                for (int r = 0; r < 8; ++r)
                    O[off + r * N] = accG[i][j][r];
            }
        }
}

// ---------------------------------------------------------------------------
// Launch
// ---------------------------------------------------------------------------
extern "C" void kernel_launch(void* const* d_in, const int* in_sizes, int n_in,
                              void* d_out, int out_size, void* d_ws, size_t ws_size,
                              hipStream_t stream) {
    const float* x    = (const float*)d_in[0];   // [2,4096,HIDDEN] f32
    const float* lut  = (const float*)d_in[1];   // [16] f32
    const int*   gidx = (const int*)d_in[2];     // [INTER,HIDDEN] i32
    const int*   uidx = (const int*)d_in[3];     // [INTER,HIDDEN] i32
    const int*   didx = (const int*)d_in[4];     // [HIDDEN,INTER] i32
    float*       out  = (float*)d_out;           // [2,4096,HIDDEN] f32

    // Workspace carve-up (f16 buffers)
    char* ws = (char*)d_ws;
    _Float16* Xh = (_Float16*)ws;  ws += (size_t)MTOK  * HIDDEN * sizeof(_Float16);
    _Float16* Wg = (_Float16*)ws;  ws += (size_t)INTER * HIDDEN * sizeof(_Float16);
    _Float16* Wu = (_Float16*)ws;  ws += (size_t)INTER * HIDDEN * sizeof(_Float16);
    _Float16* Wd = (_Float16*)ws;  ws += (size_t)HIDDEN * INTER * sizeof(_Float16);
    _Float16* Hh = (_Float16*)ws;  // [MTOK, INTER]

    // 1) dequant / convert
    {
        int n4 = (MTOK * HIDDEN) / 4;                       // 4,194,304
        f32_to_f16<<<n4 / 256, 256, 0, stream>>>(x, Xh, n4);
    }
    {
        int n4 = (INTER * HIDDEN) / 4;                      // 2,818,048
        int blk = (n4 + 255) / 256;
        lut_dequant_f16<<<blk, 256, 0, stream>>>(gidx, lut, Wg, n4);
        lut_dequant_f16<<<blk, 256, 0, stream>>>(uidx, lut, Wu, n4);
        lut_dequant_f16<<<blk, 256, 0, stream>>>(didx, lut, Wd, n4);
    }

    // 2) fused gate/up GEMM + SwiGLU -> Hh (f16), K = HIDDEN
    {
        dim3 grid(INTER / 64, MTOK / 128);                  // 86 x 64
        ghost_gemm<true><<<grid, 256, 0, stream>>>(Xh, Wg, Wu, (void*)Hh,
                                                   MTOK, INTER, HIDDEN);
    }

    // 3) down GEMM -> out (f32), K = INTER
    {
        dim3 grid(HIDDEN / 64, MTOK / 128);                 // 32 x 64
        ghost_gemm<false><<<grid, 256, 0, stream>>>(Hh, Wd, nullptr, (void*)out,
                                                    MTOK, HIDDEN, INTER);
    }
}
